// SSMLayerCuda_17102559773350
// MI455X (gfx1250) — matrix operand order repaired and checked by reading, not compiled, not run on hardware
//
#include <hip/hip_runtime.h>

// ---------------------------------------------------------------------------
// Diagonal SSM layer for MI455X (gfx1250, wave32, WMMA, async global->LDS).
//   decays = sigmoid(X*Wd + bd); inj = X*Wi + bi
//   h_t = decays_t * h_{t-1} + inj_t    (per-channel diagonal recurrence)
//   out  = LayerNorm(h)*gamma + beta + x
//
// Pipeline:
//   k0: convert x -> f16 (xh)
//   k1: convert Wd, Wi -> f16 transposed (N x K row-major)
//   k2: fused dual-GEMM (v_wmma_f32_16x16x32_f16) with DOUBLE-BUFFERED
//       async global->LDS staging (global_load_async_to_lds_b128, ASYNCcnt,
//       one barrier per K step) + bias + sigmoid + chunk-local scan epilogue
//   k3: sequential carry combine across 32 chunks per (b, d) channel
//   k4: LayerNorm fix-up + gamma/beta + residual
// ---------------------------------------------------------------------------

#define B_   8
#define T_   4096
#define D_   1024
#define M_   (B_ * T_)     // 32768
#define K_   D_
#define N_   D_
#define BM   128           // M-tile == timestep chunk length
#define BN   64            // N-tile (d channels per workgroup)
#define KB   32            // K step per WMMA iteration
#define NC   (T_ / BM)     // 32 chunks per batch
#define LN_EPS 1e-5f

#define STAGE_BYTES 16384  // per staging buffer: A 8KB + Bd 4KB + Bi 4KB

typedef _Float16 half_t;
typedef __attribute__((ext_vector_type(16))) _Float16 v16h;
typedef __attribute__((ext_vector_type(8)))  float    v8f;

union Frag {                 // 32 bytes: one 16-half WMMA operand
    v16h v;
    uint4 q[2];
};

// Async 16-byte global -> LDS copy (CDNA5 TDM-lite path, ASYNCcnt-tracked).
// lds_off: byte offset within the workgroup LDS segment (our single __shared__
// block starts at LDS offset 0). gptr: global source address.
__device__ __forceinline__ void async_copy16(unsigned lds_off, const void* gptr) {
    unsigned long long ga = (unsigned long long)gptr;
    asm volatile("global_load_async_to_lds_b128 %0, %1, off"
                 :: "v"(lds_off), "v"(ga) : "memory");
}

__device__ __forceinline__ void wait_async0() {
    asm volatile("s_wait_asynccnt 0" ::: "memory");
}

// ---------------- k0: x (f32) -> xh (f16) ----------------------------------
__global__ __launch_bounds__(256) void convert_x_kernel(
    const float* __restrict__ x, half_t* __restrict__ xh)
{
    size_t i = ((size_t)blockIdx.x * 256 + threadIdx.x) * 4;
    float4 f = *(const float4*)(x + i);
    union { half_t h[4]; uint2 u; } o;
    o.h[0] = (half_t)f.x; o.h[1] = (half_t)f.y;
    o.h[2] = (half_t)f.z; o.h[3] = (half_t)f.w;
    *(uint2*)(xh + i) = o.u;
}

// ---------------- k1: W (f32, KxN) -> Wt (f16, NxK) -------------------------
__global__ __launch_bounds__(256) void convert_w_kernel(
    const float* __restrict__ Wd, const float* __restrict__ Wi,
    half_t* __restrict__ wdt, half_t* __restrict__ wit)
{
    int idx = blockIdx.x * 256 + threadIdx.x;   // contiguous over K*N reads
    int d = idx / N_;          // K index (row of W)
    int e = idx % N_;          // N index (col of W)
    wdt[(size_t)e * K_ + d] = (half_t)Wd[idx];
    wit[(size_t)e * K_ + d] = (half_t)Wi[idx];
}

// ---------------- k2: dual GEMM + chunk-local scan --------------------------
// Grid: (M_/BM, N_/BN) = (256, 16); 256 threads = 8 waves.
// Waves arranged 4 (m) x 2 (n): each wave computes 32x32 of BOTH outputs as
// 2x2x2 WMMA accumulators.
__global__ __launch_bounds__(256) void gemm_scan_kernel(
    const half_t* __restrict__ xh,
    const half_t* __restrict__ wdt, const half_t* __restrict__ wit,
    const float* __restrict__ bd,  const float* __restrict__ bi,
    float* __restrict__ stl,   // states_local  [M_, D_]
    float* __restrict__ cup,   // cumprod decay [M_, D_]
    float* __restrict__ cS,    // chunk-final h [B_, NC, D_]
    float* __restrict__ cP)    // chunk decay prod [B_, NC, D_]
{
    // 64 KB LDS: two 16KB async staging buffers (double buffer), whole block
    // reused by the epilogue (decays 32KB + inj 32KB).
    __shared__ __align__(16) unsigned char smem[BM * BN * 4 * 2];
    float* sd = (float*)smem;                  // epilogue decays [BM][BN]
    float* si = (float*)(smem + BM * BN * 4);  // epilogue inj    [BM][BN]

    const int tid  = threadIdx.x;
    const int mt   = blockIdx.x;      // 0..255  (b = mt/32, chunk = mt%32)
    const int nt   = blockIdx.y;      // 0..15
    const int wave = tid >> 5;
    const int lane = tid & 31;
    const int lrow = lane & 15;
    const int lhi  = lane >> 4;       // lane half (ISA fragment layout)
    const int wm   = wave >> 1;       // 0..3 : rows [wm*32, wm*32+32)
    const int wn   = wave & 1;        // 0..1 : cols [wn*32, wn*32+32)

    const int m0 = mt * BM;
    const int n0 = nt * BN;

    // per-thread staging coordinates (constant across K steps)
    const int arow = tid >> 1, aseg = tid & 1;          // A: 128 rows x 2 segs
    const int t2   = tid & 127;
    const int brow = t2 >> 1, bseg = t2 & 1;            // B: 64 rows x 2 segs
    const half_t* wsrc = (tid < 128) ? wdt : wit;
    const unsigned bsub = (tid < 128) ? 0u : 4096u;     // Bd at +8KB, Bi at +12KB

    v8f accd[2][2], acci[2][2];
    {
        v8f z = {};
        for (int mi = 0; mi < 2; ++mi)
            for (int nj = 0; nj < 2; ++nj) { accd[mi][nj] = z; acci[mi][nj] = z; }
    }

    // issue async staging for K-chunk kk into staging buffer bsel
    auto issue_stage = [&](int kk, int bsel) {
        unsigned base = (unsigned)bsel * STAGE_BYTES;
        const half_t* gA = xh + (size_t)(m0 + arow) * K_ + kk + aseg * 16;
        unsigned lA = base + (unsigned)(arow * KB + aseg * 16) * 2;
        async_copy16(lA,      gA);
        async_copy16(lA + 16, gA + 8);
        const half_t* gB = wsrc + (size_t)(n0 + brow) * K_ + kk + bseg * 16;
        unsigned lB = base + 8192u + bsub + (unsigned)(brow * KB + bseg * 16) * 2;
        async_copy16(lB,      gB);
        async_copy16(lB + 16, gB + 8);
    };

    issue_stage(0, 0);   // prologue: fill buffer 0

    for (int kk = 0, it = 0; kk < K_; kk += KB, ++it) {
        const int cur = it & 1;
        // Drain this wave's outstanding async copies (the current buffer),
        // then barrier: after it, every wave's current-buffer data is in LDS
        // and every wave has finished its fragment reads of the *other*
        // buffer (compiler emits s_wait_dscnt 0 before s_barrier).
        wait_async0();
        __syncthreads();
        if (kk + KB < K_) issue_stage(kk + KB, cur ^ 1);   // prefetch next

        const half_t* Ash = (const half_t*)(smem + cur * STAGE_BYTES);
        const half_t* Bds = Ash + BM * KB;            // +4096 halves (8KB)
        const half_t* Bis = Bds + BN * KB;            // +2048 halves (4KB)

        // ---- B fragments: column n, 16 contiguous K halves per lane half ---
        Frag bf[2][2];
        for (int w = 0; w < 2; ++w)
            for (int nj = 0; nj < 2; ++nj) {
                const half_t* bp = (w ? Bis : Bds)
                                 + (wn * 32 + nj * 16 + lrow) * KB + lhi * 16;
                bf[w][nj].q[0] = *(const uint4*)bp;
                bf[w][nj].q[1] = *(const uint4*)(bp + 8);
            }
        // ---- A fragments + 8 WMMAs per K step -----------------------------
        for (int mi = 0; mi < 2; ++mi) {
            Frag af;
            const half_t* ap = Ash + (wm * 32 + mi * 16 + lrow) * KB + lhi * 16;
            af.q[0] = *(const uint4*)ap;          // K = c0 .. c0+7
            af.q[1] = *(const uint4*)(ap + 8);    // K = c0+8 .. c0+15
            accd[mi][0] = __builtin_amdgcn_wmma_f32_16x16x32_f16(
                false, af.v, false, bf[0][0].v, (short)0, accd[mi][0], false, false);
            accd[mi][1] = __builtin_amdgcn_wmma_f32_16x16x32_f16(
                false, af.v, false, bf[0][1].v, (short)0, accd[mi][1], false, false);
            acci[mi][0] = __builtin_amdgcn_wmma_f32_16x16x32_f16(
                false, af.v, false, bf[1][0].v, (short)0, acci[mi][0], false, false);
            acci[mi][1] = __builtin_amdgcn_wmma_f32_16x16x32_f16(
                false, af.v, false, bf[1][1].v, (short)0, acci[mi][1], false, false);
        }
    }

    // ---- epilogue: bias + sigmoid, deposit tiles into LDS ------------------
    __syncthreads();
    for (int mi = 0; mi < 2; ++mi)
        for (int nj = 0; nj < 2; ++nj) {
            int nloc = wn * 32 + nj * 16 + lrow;
            float bdv = bd[n0 + nloc];
            float biv = bi[n0 + nloc];
            for (int r = 0; r < 8; ++r) {
                int mloc = wm * 32 + mi * 16 + r + lhi * 8;  // C layout: M = r + 8*lhi
                float dv = accd[mi][nj][r] + bdv;
                dv = 1.0f / (1.0f + __expf(-dv));            // sigmoid
                sd[mloc * BN + nloc] = dv;
                si[mloc * BN + nloc] = acci[mi][nj][r] + biv;
            }
        }
    __syncthreads();

    // ---- chunk-local scan: 64 threads, one per d column --------------------
    if (tid < BN) {
        int dg = n0 + tid;
        int b  = mt >> 5;        // mt / (T_/BM)
        int c  = mt & 31;        // chunk index within batch
        int t0 = c * BM;
        float h = 0.0f, P = 1.0f;
        for (int t = 0; t < BM; ++t) {
            float dv = sd[t * BN + tid];
            float iv = si[t * BN + tid];
            h = fmaf(dv, h, iv);
            P *= dv;
            size_t gi = (size_t)(b * T_ + t0 + t) * D_ + dg;
            stl[gi] = h;
            cup[gi] = P;
        }
        size_t ci = (size_t)(b * NC + c) * D_ + dg;
        cS[ci] = h;
        cP[ci] = P;
    }
}

// ---------------- k3: sequential carry combine ------------------------------
__global__ __launch_bounds__(256) void carry_combine_kernel(
    const float* __restrict__ cS, const float* __restrict__ cP,
    float* __restrict__ cIn)
{
    int i = blockIdx.x * 256 + threadIdx.x;   // 0 .. B_*D_-1
    int b = i >> 10;                          // / D_
    int d = i & (D_ - 1);
    float cin = 0.0f;
    for (int c = 0; c < NC; ++c) {
        size_t idx = (size_t)(b * NC + c) * D_ + d;
        cIn[idx] = cin;
        cin = fmaf(cP[idx], cin, cS[idx]);
    }
}

// ---------------- k4: fix-up + LayerNorm + residual -------------------------
__global__ __launch_bounds__(256) void ln_kernel(
    const float* __restrict__ stl, const float* __restrict__ cup,
    const float* __restrict__ cIn, const float* __restrict__ x,
    const float* __restrict__ gamma, const float* __restrict__ beta,
    float* __restrict__ out)
{
    __shared__ float rs1[8], rs2[8];
    __shared__ float sstat[2];

    int row = blockIdx.x;                 // (b, t) flat, 0 .. M_-1
    int b   = row >> 12;                  // / T_
    int t   = row & (T_ - 1);
    int c   = t >> 7;                     // chunk (BM = 128)
    int tid = threadIdx.x;
    size_t rbase = (size_t)row * D_;
    size_t cbase = (size_t)(b * NC + c) * D_;

    float h[4];
    float s1 = 0.0f, s2 = 0.0f;
    for (int j = 0; j < 4; ++j) {
        int d = tid + j * 256;
        float v = fmaf(cIn[cbase + d], cup[rbase + d], stl[rbase + d]);
        h[j] = v;
        s1 += v;
        s2 += v * v;
    }
    // wave32 reduction, then cross-wave via LDS
    for (int off = 16; off > 0; off >>= 1) {
        s1 += __shfl_xor(s1, off, 32);
        s2 += __shfl_xor(s2, off, 32);
    }
    int wv = tid >> 5;
    if ((tid & 31) == 0) { rs1[wv] = s1; rs2[wv] = s2; }
    __syncthreads();
    if (tid == 0) {
        float a = 0.0f, q = 0.0f;
        for (int w = 0; w < 8; ++w) { a += rs1[w]; q += rs2[w]; }
        float mean = a * (1.0f / D_);
        float var  = q * (1.0f / D_) - mean * mean;
        sstat[0] = mean;
        sstat[1] = rsqrtf(var + LN_EPS);
    }
    __syncthreads();
    float mean = sstat[0], inv = sstat[1];
    for (int j = 0; j < 4; ++j) {
        int d = tid + j * 256;
        out[rbase + d] = (h[j] - mean) * inv * gamma[d] + beta[d] + x[rbase + d];
    }
}

// ---------------------------------------------------------------------------
extern "C" void kernel_launch(void* const* d_in, const int* in_sizes, int n_in,
                              void* d_out, int out_size, void* d_ws, size_t ws_size,
                              hipStream_t stream) {
    const float* x     = (const float*)d_in[0];
    const float* Wd    = (const float*)d_in[1];
    const float* bd    = (const float*)d_in[2];
    const float* Wi    = (const float*)d_in[3];
    const float* bi    = (const float*)d_in[4];
    const float* gamma = (const float*)d_in[5];
    const float* beta  = (const float*)d_in[6];
    float* out = (float*)d_out;

    // workspace carve (~343 MB total)
    char* ws = (char*)d_ws;
    half_t* xh  = (half_t*)ws;  ws += (size_t)M_ * K_ * 2;     // 64 MB
    half_t* wdt = (half_t*)ws;  ws += (size_t)K_ * N_ * 2;     // 2 MB
    half_t* wit = (half_t*)ws;  ws += (size_t)K_ * N_ * 2;     // 2 MB
    float*  stl = (float*)ws;   ws += (size_t)M_ * D_ * 4;     // 128 MB
    float*  cup = (float*)ws;   ws += (size_t)M_ * D_ * 4;     // 128 MB
    float*  cS  = (float*)ws;   ws += (size_t)B_ * NC * D_ * 4;
    float*  cP  = (float*)ws;   ws += (size_t)B_ * NC * D_ * 4;
    float*  cIn = (float*)ws;   ws += (size_t)B_ * NC * D_ * 4;

    convert_x_kernel<<<(M_ * K_) / (4 * 256), 256, 0, stream>>>(x, xh);
    convert_w_kernel<<<(K_ * N_) / 256, 256, 0, stream>>>(Wd, Wi, wdt, wit);

    dim3 g(M_ / BM, N_ / BN);   // (256, 16)
    gemm_scan_kernel<<<g, 256, 0, stream>>>(xh, wdt, wit, bd, bi,
                                            stl, cup, cS, cP);

    carry_combine_kernel<<<(B_ * D_) / 256, 256, 0, stream>>>(cS, cP, cIn);

    ln_kernel<<<M_, 256, 0, stream>>>(stl, cup, cIn, x, gamma, beta, out);
}
